// WindowAttentionMemory_69080253988942
// MI455X (gfx1250) — matrix operand order repaired and checked by reading, not compiled, not run on hardware
//
#include <hip/hip_runtime.h>
#include <hip/hip_fp16.h>
#include <stdint.h>

// Problem constants (match reference)
#define BB 4
#define SS 2048
#define EE 1024
#define HH 16
#define DD 64
#define WW 256
#define NROW (BB * SS)   // 8192 token rows

typedef _Float16 half_t;
typedef half_t v16h __attribute__((ext_vector_type(16)));
typedef half_t v8h  __attribute__((ext_vector_type(8)));
typedef half_t v4h  __attribute__((ext_vector_type(4)));
typedef float  v8f  __attribute__((ext_vector_type(8)));

// ---- workspace layout (units: half_t elements) ----
static constexpr size_t OFF_XN = 0;                               // xn   [8192,1024] f16
static constexpr size_t OFF_W  = OFF_XN + (size_t)NROW * EE;      // wq,wk,wv,wo f16, each [1024,1024]
static constexpr size_t OFF_Q  = OFF_W  + (size_t)4 * EE * EE;    // Q [B,H,S,D] f16
static constexpr size_t OFF_K  = OFF_Q  + (size_t)NROW * EE;      // K
static constexpr size_t OFF_V  = OFF_K  + (size_t)NROW * EE;      // V
static constexpr size_t OFF_C  = OFF_V  + (size_t)NROW * EE;      // ctx [8192,1024] f16

__device__ __forceinline__ v16h cat8(v8h a, v8h b) {
  return __builtin_shufflevector(a, b, 0,1,2,3,4,5,6,7,8,9,10,11,12,13,14,15);
}

__device__ __forceinline__ v8f wmma_f16(v16h a, v16h b, v8f c) {
  // D = A(16x32) * B(32x16) + C, f32 accumulate
  return __builtin_amdgcn_wmma_f32_16x16x32_f16(false, a, false, b, (short)0, c, false, false);
}

// =====================================================================
// Kernel 1: LayerNorm (one block per token row) + f32 -> f16 convert
// =====================================================================
__global__ void __launch_bounds__(256) ln_kernel(const float* __restrict__ x,
                                                 const float* __restrict__ g,
                                                 const float* __restrict__ be,
                                                 half_t* __restrict__ xn) {
  const int row = blockIdx.x;
  const int t   = threadIdx.x;
  const float4 v = ((const float4*)(x + (size_t)row * EE))[t];
  float s1 = v.x + v.y + v.z + v.w;
  float s2 = v.x*v.x + v.y*v.y + v.z*v.z + v.w*v.w;
#pragma unroll
  for (int m = 16; m >= 1; m >>= 1) {
    s1 += __shfl_xor(s1, m, 32);
    s2 += __shfl_xor(s2, m, 32);
  }
  __shared__ float red1[8], red2[8];
  const int wave = t >> 5, lane = t & 31;
  if (lane == 0) { red1[wave] = s1; red2[wave] = s2; }
  __syncthreads();
  float ts1 = 0.f, ts2 = 0.f;
#pragma unroll
  for (int i = 0; i < 8; ++i) { ts1 += red1[i]; ts2 += red2[i]; }
  const float mu  = ts1 * (1.0f / EE);
  const float var = ts2 * (1.0f / EE) - mu * mu;
  const float rs  = rsqrtf(var + 1e-12f);
  const float4 gg = ((const float4*)g)[t];
  const float4 bb = ((const float4*)be)[t];
  v4h o;
  o[0] = (half_t)((v.x - mu) * rs * gg.x + bb.x);
  o[1] = (half_t)((v.y - mu) * rs * gg.y + bb.y);
  o[2] = (half_t)((v.z - mu) * rs * gg.z + bb.z);
  o[3] = (half_t)((v.w - mu) * rs * gg.w + bb.w);
  *(v4h*)(xn + (size_t)row * EE + t * 4) = o;
}

// =====================================================================
// Kernel 2: weight f32 -> f16 convert (vectorized)
// =====================================================================
__global__ void __launch_bounds__(256) cvt_kernel(const float* __restrict__ src,
                                                  half_t* __restrict__ dst, int n4) {
  int i = blockIdx.x * blockDim.x + threadIdx.x;
  if (i < n4) {
    float4 v = ((const float4*)src)[i];
    v4h o;
    o[0] = (half_t)v.x; o[1] = (half_t)v.y; o[2] = (half_t)v.z; o[3] = (half_t)v.w;
    ((v4h*)dst)[i] = o;
  }
}

// =====================================================================
// Kernel 3: QKV projection GEMM.  y = xn @ W^T + b  (W is [f,e] row-major,
// so B-fragments are contiguous row reads of W -- free transpose).
// One wave computes a 16(M) x 64(N) output tile; K-loop over E in steps of 32.
// Output written as f16 in [B,H,S,D] layout.
// =====================================================================
__global__ void __launch_bounds__(128) proj_kernel(const half_t* __restrict__ xn,
                                                   const half_t* __restrict__ wbase,
                                                   const float* __restrict__ bq,
                                                   const float* __restrict__ bk,
                                                   const float* __restrict__ bvp,
                                                   half_t* __restrict__ qh,
                                                   half_t* __restrict__ kh,
                                                   half_t* __restrict__ vh) {
  const int wave = threadIdx.x >> 5, lane = threadIdx.x & 31;
  const int hw = lane >> 4, lr = lane & 15;
  const int row0 = (blockIdx.x * 4 + wave) * 16;
  const int n0   = blockIdx.y * 64;
  const int z    = blockIdx.z;

  const half_t* Wm  = wbase + (size_t)z * EE * EE;
  const float* bias = (z == 0) ? bq : (z == 1) ? bk : bvp;
  half_t* outp      = (z == 0) ? qh : (z == 1) ? kh : vh;

  const half_t* arow = xn + (size_t)(row0 + lr) * EE;

  v8f zero = {};
  v8f acc[4] = {zero, zero, zero, zero};

  for (int k0 = 0; k0 < EE; k0 += 32) {
    v16h af = *(const v16h*)(arow + k0 + 16 * hw);
#pragma unroll
    for (int t = 0; t < 4; ++t) {
      const half_t* bp = Wm + (size_t)(n0 + t * 16 + lr) * EE + k0 + 16 * hw;
      v16h bf = *(const v16h*)bp;
      acc[t] = wmma_f16(af, bf, acc[t]);
    }
  }

  const int b = row0 >> 11;  // / SS
#pragma unroll
  for (int t = 0; t < 4; ++t) {
    const int f  = n0 + t * 16 + lr;
    const int hh = f >> 6, d = f & 63;
    const float bval = bias[f];
#pragma unroll
    for (int r = 0; r < 8; ++r) {
      const int rw = row0 + r + 8 * hw;
      const int s  = rw & (SS - 1);
      const size_t dst = ((size_t)(b * HH + hh) * SS + s) * DD + d;
      outp[dst] = (half_t)(acc[t][r] + bval);
    }
  }
}

// =====================================================================
// Kernel 4: sliding-window causal flash attention.
// One wave owns one (b,h,16-query tile). Streams 32-key blocks; V is staged
// to LDS with gfx1250 async global->LDS copies, transposed in LDS to build
// WMMA B-fragments. P round-trips through LDS (C-layout -> A-layout).
// =====================================================================
__global__ void __launch_bounds__(128) attn_kernel(const half_t* __restrict__ qh,
                                                   const half_t* __restrict__ kh,
                                                   const half_t* __restrict__ vh,
                                                   half_t* __restrict__ cx) {
  // per-wave LDS: Vrow[32x64] + Vt[64x32] + P[16x32] = 4608 halfs = 9216 B
  __shared__ __align__(32) half_t smem[4 * 4608];

  const int wave = threadIdx.x >> 5, lane = threadIdx.x & 31;
  const int hw = lane >> 4, lr = lane & 15;

  half_t* Vrow = smem + wave * 4608;
  half_t* Vt   = Vrow + 2048;
  half_t* Pt   = Vt + 2048;

  const int wid = blockIdx.x * 4 + wave;
  const int qt  = wid & (SS / 16 - 1);
  const int bh  = wid >> 7;          // / (SS/16)
  const int h   = bh & (HH - 1);
  const int b   = bh >> 4;
  const int qs  = qt * 16;
  const size_t bh_off = (size_t)(b * HH + h) * SS;

  // Q A-fragments (rows qs..qs+15, d split into two K=32 chunks)
  const half_t* qrow = qh + (bh_off + qs + lr) * DD;
  v16h aq[2];
#pragma unroll
  for (int i = 0; i < 2; ++i) {
    aq[i] = *(const v16h*)(qrow + i * 32 + 16 * hw);
  }

  v8f zero = {};
  v8f acc[4] = {zero, zero, zero, zero};
  float mrow[8], lrow[8];
#pragma unroll
  for (int r = 0; r < 8; ++r) { mrow[r] = 0.f; lrow[r] = 0.f; }

  int lo = qs - (WW - 1);
  if (lo < 0) lo = 0;
  const int kb_start = lo & ~31;
  const float scale = 0.125f;  // 1/sqrt(64)

  for (int kb = kb_start; kb <= qs + 15; kb += 32) {
    // ---- issue async stage of V[kb..kb+31, 0..63] into LDS (row-major) ----
    asm volatile("s_wait_dscnt 0" ::: "memory");  // prior-iter LDS reads done
    const half_t* vbase = vh + (bh_off + kb) * DD;
#pragma unroll
    for (int c = 0; c < 8; ++c) {
      const int id = c * 32 + lane;          // 256 chunks of 16B
      const int rrow = id >> 3, part = id & 7;
      uint32_t ldst = (uint32_t)(uintptr_t)(Vrow + rrow * DD + part * 8);
      uint64_t gsrc = (uint64_t)(uintptr_t)(vbase + rrow * DD + part * 8);
      asm volatile("global_load_async_to_lds_b128 %0, %1, off"
                   :: "v"(ldst), "v"(gsrc) : "memory");
    }

    // ---- scores: S = Q . K^T (overlaps with async V copy) ----
    const half_t* kp0 = kh + (bh_off + kb + lr) * DD;
    const half_t* kp1 = kh + (bh_off + kb + 16 + lr) * DD;
    v8f sc0 = {}, sc1 = {};
    {
      v16h b00 = *(const v16h*)(kp0 + 16 * hw);
      v16h b01 = *(const v16h*)(kp0 + 32 + 16 * hw);
      sc0 = wmma_f16(aq[0], b00, sc0);
      sc0 = wmma_f16(aq[1], b01, sc0);
      v16h b10 = *(const v16h*)(kp1 + 16 * hw);
      v16h b11 = *(const v16h*)(kp1 + 32 + 16 * hw);
      sc1 = wmma_f16(aq[0], b10, sc1);
      sc1 = wmma_f16(aq[1], b11, sc1);
    }

    // ---- mask + scale (C layout: lane(hw,lr), VGPR r -> M=r+8*hw, N=lr) ----
#pragma unroll
    for (int r = 0; r < 8; ++r) {
      const int qr = qs + r + 8 * hw;
      const int k0i = kb + lr, k1i = kb + 16 + lr;
      sc0[r] = ((qr >= k0i) && (qr - k0i < WW)) ? sc0[r] * scale : -1e30f;
      sc1[r] = ((qr >= k1i) && (qr - k1i < WW)) ? sc1[r] * scale : -1e30f;
    }

    // ---- wait for async V, transpose in LDS: Vt[d][key] = Vrow[key][d] ----
    asm volatile("s_wait_asynccnt 0" ::: "memory");
#pragma unroll
    for (int c = 0; c < 8; ++c) {
      const int id = c * 32 + lane;
      const int rrow = id >> 3, part = id & 7;
      v8h chunk = *(const v8h*)(Vrow + rrow * DD + part * 8);
#pragma unroll
      for (int j = 0; j < 8; ++j)
        Vt[(part * 8 + j) * 32 + rrow] = chunk[j];
    }

    // ---- online softmax (row stats live per half-wave) + write P to LDS ----
#pragma unroll
    for (int r = 0; r < 8; ++r) {
      float red = fmaxf(sc0[r], sc1[r]);
      red = fmaxf(red, __shfl_xor(red, 1, 32));
      red = fmaxf(red, __shfl_xor(red, 2, 32));
      red = fmaxf(red, __shfl_xor(red, 4, 32));
      red = fmaxf(red, __shfl_xor(red, 8, 32));
      const float nm = fmaxf(mrow[r], red);
      const float alpha = __expf(mrow[r] - nm);
      const float p0 = __expf(sc0[r] - nm);
      const float p1 = __expf(sc1[r] - nm);
      float ps = p0 + p1;
      ps += __shfl_xor(ps, 1, 32);
      ps += __shfl_xor(ps, 2, 32);
      ps += __shfl_xor(ps, 4, 32);
      ps += __shfl_xor(ps, 8, 32);
      lrow[r] = lrow[r] * alpha + ps;
      mrow[r] = nm;
#pragma unroll
      for (int dt = 0; dt < 4; ++dt) acc[dt][r] *= alpha;
      const int prow = r + 8 * hw;
      Pt[prow * 32 + lr]      = (half_t)p0;
      Pt[prow * 32 + 16 + lr] = (half_t)p1;
    }

    // ---- P A-fragment (16x32 over keys) ----
    v16h pa = *(const v16h*)(Pt + lr * 32 + 16 * hw);

    // ---- PV: acc += P(16x32) * V(32x16) for 4 d-subtiles ----
#pragma unroll
    for (int dt = 0; dt < 4; ++dt) {
      v16h bv = *(const v16h*)(Vt + (dt * 16 + lr) * 32 + 16 * hw);
      acc[dt] = wmma_f16(pa, bv, acc[dt]);
    }
  }

  // ---- normalize + store ctx as f16 [row, e=h*64+d] ----
  const int crow = b * SS + qs;
#pragma unroll
  for (int dt = 0; dt < 4; ++dt) {
#pragma unroll
    for (int r = 0; r < 8; ++r) {
      const int m = r + 8 * hw;
      const size_t dst = (size_t)(crow + m) * EE + h * DD + dt * 16 + lr;
      cx[dst] = (half_t)(acc[dt][r] / lrow[r]);
    }
  }
}

// =====================================================================
// Kernel 5: output projection + bias + residual (f32 output).
// =====================================================================
__global__ void __launch_bounds__(128) oproj_kernel(const half_t* __restrict__ cx,
                                                    const half_t* __restrict__ wo,
                                                    const float* __restrict__ bo,
                                                    const float* __restrict__ x,
                                                    float* __restrict__ out) {
  const int wave = threadIdx.x >> 5, lane = threadIdx.x & 31;
  const int hw = lane >> 4, lr = lane & 15;
  const int row0 = (blockIdx.x * 4 + wave) * 16;
  const int n0   = blockIdx.y * 64;

  const half_t* arow = cx + (size_t)(row0 + lr) * EE;

  v8f zero = {};
  v8f acc[4] = {zero, zero, zero, zero};

  for (int k0 = 0; k0 < EE; k0 += 32) {
    v16h af = *(const v16h*)(arow + k0 + 16 * hw);
#pragma unroll
    for (int t = 0; t < 4; ++t) {
      const half_t* bp = wo + (size_t)(n0 + t * 16 + lr) * EE + k0 + 16 * hw;
      v16h bf = *(const v16h*)bp;
      acc[t] = wmma_f16(af, bf, acc[t]);
    }
  }

#pragma unroll
  for (int t = 0; t < 4; ++t) {
    const int f = n0 + t * 16 + lr;
    const float bval = bo[f];
#pragma unroll
    for (int r = 0; r < 8; ++r) {
      const int rw = row0 + r + 8 * hw;
      const size_t idx = (size_t)rw * EE + f;
      out[idx] = x[idx] + acc[t][r] + bval;
    }
  }
}

// =====================================================================
extern "C" void kernel_launch(void* const* d_in, const int* in_sizes, int n_in,
                              void* d_out, int out_size, void* d_ws, size_t ws_size,
                              hipStream_t stream) {
  (void)in_sizes; (void)n_in; (void)out_size; (void)ws_size;
  const float* x   = (const float*)d_in[0];
  const float* lng = (const float*)d_in[1];
  const float* lnb = (const float*)d_in[2];
  const float* wq  = (const float*)d_in[3];
  const float* bq  = (const float*)d_in[4];
  const float* wk  = (const float*)d_in[5];
  const float* bk  = (const float*)d_in[6];
  const float* wv  = (const float*)d_in[7];
  const float* bv  = (const float*)d_in[8];
  const float* wo  = (const float*)d_in[9];
  const float* bo  = (const float*)d_in[10];
  float* out = (float*)d_out;

  half_t* ws = (half_t*)d_ws;
  half_t* xn = ws + OFF_XN;
  half_t* wb = ws + OFF_W;
  half_t* qh = ws + OFF_Q;
  half_t* kh = ws + OFF_K;
  half_t* vh = ws + OFF_V;
  half_t* cx = ws + OFF_C;

  // 1) LayerNorm + f16 convert
  ln_kernel<<<NROW, 256, 0, stream>>>(x, lng, lnb, xn);

  // 2) weight converts
  const int n4 = EE * EE / 4;
  const int cb = (n4 + 255) / 256;
  cvt_kernel<<<cb, 256, 0, stream>>>(wq, wb + (size_t)0 * EE * EE, n4);
  cvt_kernel<<<cb, 256, 0, stream>>>(wk, wb + (size_t)1 * EE * EE, n4);
  cvt_kernel<<<cb, 256, 0, stream>>>(wv, wb + (size_t)2 * EE * EE, n4);
  cvt_kernel<<<cb, 256, 0, stream>>>(wo, wb + (size_t)3 * EE * EE, n4);

  // 3) QKV projections (WMMA)
  proj_kernel<<<dim3(NROW / 64, EE / 64, 3), 128, 0, stream>>>(
      xn, wb, bq, bk, bv, qh, kh, vh);

  // 4) sliding-window flash attention (WMMA + async LDS staging)
  attn_kernel<<<BB * HH * (SS / 16) / 4, 128, 0, stream>>>(qh, kh, vh, cx);

  // 5) output projection + residual (WMMA)
  oproj_kernel<<<dim3(NROW / 64, EE / 64), 128, 0, stream>>>(
      cx, wb + (size_t)3 * EE * EE, bo, x, out);
}
